// SDFHashGridNetwork_69612829933842
// MI455X (gfx1250) — matrix operand order repaired and checked by reading, not compiled, not run on hardware
//
#include <hip/hip_runtime.h>
#include <math.h>

typedef _Float16 f16;
typedef __attribute__((ext_vector_type(16))) _Float16 v16h;
typedef __attribute__((ext_vector_type(8)))  float    v8f;

#define HG_LOG2_T 19
#define HG_T (1u << HG_LOG2_T)

// scale_l = 16 * 128^(l/15) - 1 (tcnn growth, PER_LEVEL_SCALE = 2^(7/15))
__device__ __constant__ float kScaleLo[8] = {
    15.0f, 21.1106058f, 29.5549264f, 41.2242464f,
    57.3502192f, 79.6349184f, 110.430437f, 152.987100f};
__device__ __constant__ float kScaleHi[8] = {
    211.796755f, 293.066584f, 405.374379f, 560.573992f,
    775.046192f, 1071.42821f, 1481.00184f, 2047.0f};
// res = ceil(scale)+1 for the dense levels (levels 0..4; 59^3 <= 2^19 < 81^3)
__device__ __constant__ int kResDense[5] = {16, 23, 31, 43, 59};

// ---------------- WMMA A loader (wave32, 16x16x32 f16) ----------------
// A matrix 16x32 f16 (MxK) from row-major [16][64] LDS tile.
// Contiguous K order (matches B packing): lanes 0-15: V0..V7 -> K=0..15
//                                        lanes16-31: V0..V7 -> K=16..31
__device__ __forceinline__ v16h load_A(const f16* __restrict__ buf, int lane, int kbase) {
  const int M  = lane & 15;
  const int h8 = (lane >> 4) * 16;
  const f16* row = buf + M * 64 + kbase;
  v16h a;
#pragma unroll
  for (int v = 0; v < 8; ++v) {
    const int k = h8 + 2 * v;
    a[2 * v + 0] = row[k + 0];
    a[2 * v + 1] = row[k + 1];
  }
  return a;
}

// Branch-free Softplus(beta=100): 0.01*(max(t,0) + ln(1+exp(-|t|))), t=100x.
// Uses native v_exp_f32/v_log_f32 (base-2) transcendentals.
__device__ __forceinline__ float softplus100(float x) {
  const float t = 100.0f * x;
  const float u = __builtin_amdgcn_exp2f(-fabsf(t) * 1.44269504088896341f);
  const float l = __builtin_amdgcn_logf(1.0f + u) * 0.693147180559945310f;
  return (fmaxf(t, 0.0f) + l) * 0.01f;
}

// -----------------------------------------------------------------------------
// Fused: hashgrid encode (16 levels, trilerp, dense/hash) -> 35->64 softplus
// -> 64->13, per 128-point block. 8 waves x 16 points. Weights are weight-
// normed once per block and packed straight into WMMA B-fragment layout in LDS.
// -----------------------------------------------------------------------------
__global__ __launch_bounds__(256)
void sdf_hashgrid_mlp_kernel(const float* __restrict__ x,      // [N,3]
                             const float* __restrict__ table,  // [16,T,2]
                             const float* __restrict__ v1,     // [64,35]
                             const float* __restrict__ g1,     // [64]
                             const float* __restrict__ b1,     // [64]
                             const float* __restrict__ v2,     // [13,64]
                             const float* __restrict__ g2,     // [13]
                             const float* __restrict__ b2,     // [13]
                             float* __restrict__ out,          // [N,13]
                             int n) {
  // B fragments: [frag][lane][16 halves] -> one aligned 32B LDS load per lane.
  // w1: frag = nt*2 + kt (4 col-tiles x 2 k-tiles); w2: frag = kt.
  __shared__ __align__(32) f16 w1f[8][32][16];  // 8192 B
  __shared__ __align__(32) f16 w2f[2][32][16];  // 2048 B
  __shared__ float b1s[64];
  __shared__ float b2s[16];
  __shared__ float g1inv[64];
  __shared__ float g2inv[16];
  __shared__ __align__(32) f16 hbuf[8][16 * 64];    // per-wave input  [pt][K]
  __shared__ __align__(32) f16 hidbuf[8][16 * 64];  // per-wave hidden [pt][K]

  const int t    = threadIdx.x;
  const int wave = t >> 5;
  const int lane = t & 31;
  const int pl   = lane & 15;  // point within wave tile
  const int hi   = lane >> 4;  // lane half (levels 0-7 vs 8-15)

  // ---- zero input staging (K=35..63 must stay zero) ----
  {
    uint32_t* z = (uint32_t*)&hbuf[0][0];  // 8*1024 halves = 4096 dwords
#pragma unroll
    for (int i = 0; i < 16; ++i) z[t + 256 * i] = 0u;
  }

  // ---- weight_norm phase 1: per-row g/||v|| scales ----
  if (t < 64) {
    float nrm = 0.f;
#pragma unroll
    for (int k = 0; k < 35; ++k) { const float v = v1[t * 35 + k]; nrm += v * v; }
    g1inv[t] = g1[t] / sqrtf(nrm);
    b1s[t]   = b1[t];
  }
  if (t < 16) {
    if (t < 13) {
      float nrm = 0.f;
#pragma unroll
      for (int k = 0; k < 64; ++k) { const float v = v2[t * 64 + k]; nrm += v * v; }
      g2inv[t] = g2[t] / sqrtf(nrm);
      b2s[t]   = b2[t];
    } else {
      g2inv[t] = 0.f;
      b2s[t]   = 0.f;
    }
  }
  __syncthreads();

  // ---- weight_norm phase 2: pack W1/W2 into B-fragment layout ----
  {
    // 8 frags x 32 lane-slots = 256 entries == 256 threads, 16 halves each.
    const int fi  = t >> 5;                       // frag id: nt = fi>>1, kt = fi&1
    const int fl  = t & 31;                       // lane slot
    const int col = (fi >> 1) * 16 + (fl & 15);   // hidden unit j
    const int kb  = (fi & 1) * 32 + (fl >> 4) * 16;
    const float sc = g1inv[col];
    f16* dst = &w1f[fi][fl][0];
#pragma unroll
    for (int e = 0; e < 16; ++e) {
      const int k = kb + e;
      dst[e] = (k < 35) ? (f16)(v1[col * 35 + k] * sc) : (f16)0.f;
    }
  }
  if (t < 64) {  // 2 frags x 32 lane-slots
    const int fi = t >> 5;
    const int fl = t & 31;
    const int j  = fl & 15;
    const int kb = fi * 32 + (fl >> 4) * 16;
    const float sc = (j < 13) ? g2inv[j] : 0.f;
    f16* dst = &w2f[fi][fl][0];
#pragma unroll
    for (int e = 0; e < 16; ++e)
      dst[e] = (j < 13) ? (f16)(v2[j * 64 + kb + e] * sc) : (f16)0.f;
  }

  // ---- hashgrid encoding: 2 lanes per point, 8 levels per lane ----
  const int p  = blockIdx.x * 128 + wave * 16 + pl;
  const int pc = (p < n) ? p : (n - 1);
  const float xr = __builtin_nontemporal_load(&x[pc * 3 + 0]);
  const float yr = __builtin_nontemporal_load(&x[pc * 3 + 1]);
  const float zr = __builtin_nontemporal_load(&x[pc * 3 + 2]);
  const float x0 = fminf(fmaxf(xr + 0.5f, 0.f), 1.f);
  const float y0 = fminf(fmaxf(yr + 0.5f, 0.f), 1.f);
  const float z0 = fminf(fmaxf(zr + 0.5f, 0.f), 1.f);

  f16* hrow = &hbuf[wave][pl * 64];
  if (hi == 0) {  // raw xyz -> columns 0..2
    hrow[0] = (f16)xr;
    hrow[1] = (f16)yr;
    hrow[2] = (f16)zr;
  }

#pragma unroll
  for (int li = 0; li < 8; ++li) {
    const int   lvl   = hi * 8 + li;
    const float scale = hi ? kScaleHi[li] : kScaleLo[li];  // compile-time idx

    const float px = x0 * scale + 0.5f;
    const float py = y0 * scale + 0.5f;
    const float pz = z0 * scale + 0.5f;
    const float fx = floorf(px), fy = floorf(py), fz = floorf(pz);
    const float wx = px - fx, wy = py - fy, wz = pz - fz;
    const uint32_t cx0 = (uint32_t)fx, cy0 = (uint32_t)fy, cz0 = (uint32_t)fz;

    const float2* tab = (const float2*)table + ((size_t)lvl << HG_LOG2_T);
    float a0 = 0.f, a1 = 0.f;
#pragma unroll
    for (int c = 0; c < 8; ++c) {
      const uint32_t ox = (uint32_t)(c & 1), oy = (uint32_t)((c >> 1) & 1),
                     oz = (uint32_t)((c >> 2) & 1);
      const uint32_t cx = cx0 + ox, cy = cy0 + oy, cz = cz0 + oz;
      uint32_t idx;
      if (li < 5 && hi == 0) {  // dense levels (0..4) live only in lane-half 0
        const uint32_t res = (uint32_t)kResDense[li < 5 ? li : 0];
        idx = cx + cy * res + cz * res * res;
      } else {
        idx = (cx * 1u) ^ (cy * 2654435761u) ^ (cz * 805459861u);
        idx &= (HG_T - 1u);
      }
      const float2 f = tab[idx];  // L2-resident gather (64 MB table < 192 MB L2)
      const float wc = (ox ? wx : 1.f - wx) * (oy ? wy : 1.f - wy) * (oz ? wz : 1.f - wz);
      a0 += wc * f.x;
      a1 += wc * f.y;
    }
    hrow[3 + lvl * 2 + 0] = (f16)a0;
    hrow[3 + lvl * 2 + 1] = (f16)a1;
  }
  __syncthreads();

  // ---- layer 1: [16x64] = softplus([16x64(K)] x [64x64] + b1), via WMMA ----
  const f16* hb  = &hbuf[wave][0];
  f16*       hob = &hidbuf[wave][0];
  const v16h A0  = load_A(hb, lane, 0);
  const v16h A1  = load_A(hb, lane, 32);

#pragma unroll
  for (int nt = 0; nt < 4; ++nt) {
    const v16h B0 = *(const v16h*)&w1f[nt * 2 + 0][lane][0];
    const v16h B1 = *(const v16h*)&w1f[nt * 2 + 1][lane][0];
    const float bj = b1s[nt * 16 + pl];  // C/D column = hidden unit
    v8f acc = {bj, bj, bj, bj, bj, bj, bj, bj};
    acc = __builtin_amdgcn_wmma_f32_16x16x32_f16(false, A0, false, B0, (short)0,
                                                 acc, false, false);
    acc = __builtin_amdgcn_wmma_f32_16x16x32_f16(false, A1, false, B1, (short)0,
                                                 acc, false, false);
#pragma unroll
    for (int r = 0; r < 8; ++r) {  // C/D: VGPR r -> row M = r + 8*hi
      const int M = r + hi * 8;
      hob[M * 64 + nt * 16 + pl] = (f16)softplus100(acc[r]);
    }
  }
  __syncthreads();

  // ---- layer 2: [16x16] = [16x64] x [64x16] + b2, via WMMA ----
  const v16h H0 = load_A(hob, lane, 0);
  const v16h H1 = load_A(hob, lane, 32);
  const v16h C0 = *(const v16h*)&w2f[0][lane][0];
  const v16h C1 = *(const v16h*)&w2f[1][lane][0];
  const float cj = b2s[pl];
  v8f o = {cj, cj, cj, cj, cj, cj, cj, cj};
  o = __builtin_amdgcn_wmma_f32_16x16x32_f16(false, H0, false, C0, (short)0,
                                             o, false, false);
  o = __builtin_amdgcn_wmma_f32_16x16x32_f16(false, H1, false, C1, (short)0,
                                             o, false, false);

  // ---- store [N,13] non-temporally (keep the table resident in L2) ----
  if (pl < 13) {
#pragma unroll
    for (int r = 0; r < 8; ++r) {
      const int M  = r + hi * 8;
      const int pg = blockIdx.x * 128 + wave * 16 + M;
      if (pg < n) __builtin_nontemporal_store(o[r], &out[(size_t)pg * 13 + pl]);
    }
  }
}

extern "C" void kernel_launch(void* const* d_in, const int* in_sizes, int n_in,
                              void* d_out, int out_size, void* d_ws, size_t ws_size,
                              hipStream_t stream) {
  (void)n_in; (void)out_size; (void)d_ws; (void)ws_size;
  const float* x     = (const float*)d_in[0];
  const float* table = (const float*)d_in[1];
  const float* v1    = (const float*)d_in[2];
  const float* g1    = (const float*)d_in[3];
  const float* b1    = (const float*)d_in[4];
  const float* v2    = (const float*)d_in[5];
  const float* g2    = (const float*)d_in[6];
  const float* b2    = (const float*)d_in[7];
  float*       out   = (float*)d_out;

  const int n      = in_sizes[0] / 3;     // N = 2^21
  const int blocks = (n + 127) / 128;     // 128 points per block (8 waves x 16)
  sdf_hashgrid_mlp_kernel<<<blocks, 256, 0, stream>>>(x, table, v1, g1, b1, v2,
                                                      g2, b2, out, n);
}
